// RNN_38087769981203
// MI455X (gfx1250) — compile-verified, matrix operand order checked by reading, not run on hardware
//
#include <hip/hip_runtime.h>
#include <hip/hip_bf16.h>

typedef __attribute__((ext_vector_type(16))) _Float16 v16h;
typedef __attribute__((ext_vector_type(8)))  _Float16 v8h;
typedef __attribute__((ext_vector_type(8)))  float    v8f;

#define RNN_B   32
#define RNN_T   16384
#define RNN_H   64
#define RNN_G   192          // 3*H
#define EULER_K 0.5f
#define ROWP    72           // padded row stride (halves) for LDS h tile: 16B pad kills bank conflicts

// Hardware v_rcp_f32 (~1 ulp) instead of the IEEE div_scale/div_fmas/div_fixup
// sequence: keeps the 16384-step serial chain short.
__device__ __forceinline__ float fast_rcp(float v) {
    return __builtin_amdgcn_rcpf(v);
}
__device__ __forceinline__ float fast_sigmoid(float v) {
    return fast_rcp(1.0f + __expf(-v));
}
__device__ __forceinline__ float fast_tanh(float v) {
    float e = __expf(2.0f * v);          // saturates correctly at +/-inf
    return 1.0f - 2.0f * fast_rcp(e + 1.0f);
}

// Euler-GRU recurrence. One block = 16 batch rows, 4 waves; wave w owns hidden
// channels [16w, 16w+16) across all three gates. w_hh lives in registers (f16,
// WMMA B layout). h double-buffered in LDS (f16, A-operand source) + per-lane
// f32 registers for the elementwise update (C layout == update layout).
// x is software-pipelined one step ahead so its global latency never sits on
// the serial chain.
__global__ void __launch_bounds__(128)
gru_scan_kernel(const float* __restrict__ x,
                const float* __restrict__ w_ih,
                const float* __restrict__ w_hh,
                const float* __restrict__ b_ih,
                const float* __restrict__ b_hh,
                float* __restrict__ states)   // [B,T,H]
{
    __shared__ __align__(16) _Float16 hbuf[2][16 * ROWP];

    const int tid  = threadIdx.x;
    const int wave = tid >> 5;
    const int lane = tid & 31;
    const int half = lane >> 4;          // 0: lanes 0-15, 1: lanes 16-31
    const int lq   = lane & 15;
    const int ch   = wave * 16 + lq;     // hidden channel this lane owns (N index)
    const int b0   = blockIdx.x * 16;    // batch tile base

    // ---- per-lane scalar weights/biases (gate rows: r=ch, z=64+ch, n=128+ch) ----
    const float wir  = w_ih[ch];
    const float wiz  = w_ih[64 + ch];
    const float win  = w_ih[128 + ch];
    const float br   = b_ih[ch]       + b_hh[ch];
    const float bz   = b_ih[64 + ch]  + b_hh[64 + ch];
    const float bihn = b_ih[128 + ch];
    const float bhhn = b_hh[128 + ch];

    // ---- preload w_hh into registers in WMMA B layout (f16) ----
    // B (32x16, K x N): lane holds N = lq; K-runs of 8: lanes 0-15 -> {K0+0..7, K0+16..23},
    // lanes 16-31 -> {K0+8..15, K0+24..31}. gh = h @ w_hh^T  =>  B[k][n] = w_hh[row(n)][k].
    v16h bm[3][2];
    #pragma unroll
    for (int g = 0; g < 3; ++g) {
        const float* wrow = w_hh + (size_t)(g * 64 + ch) * 64;
        #pragma unroll
        for (int kc = 0; kc < 2; ++kc) {
            const int kf = kc * 32 + half * 8;
            v16h tmp;
            #pragma unroll
            for (int i = 0; i < 8; ++i) tmp[i]     = (_Float16)wrow[kf + i];
            #pragma unroll
            for (int i = 0; i < 8; ++i) tmp[8 + i] = (_Float16)wrow[kf + 16 + i];
            bm[g][kc] = tmp;
        }
    }

    // ---- zero initial h (LDS buffer 0 + registers) ----
    for (int i = tid; i < 2 * 16 * ROWP; i += blockDim.x)
        hbuf[0][i] = (_Float16)0.0f;     // covers both buffers (flat indexing)
    float hreg[8];
    #pragma unroll
    for (int j = 0; j < 8; ++j) hreg[j] = 0.0f;
    __syncthreads();

    const v8f zero = {0.f, 0.f, 0.f, 0.f, 0.f, 0.f, 0.f, 0.f};
    const size_t xrow = (size_t)(b0 + lq) * RNN_T;

    // Prefetch x for t=0 (consumed in iteration 0; each iteration prefetches t+1).
    float xpref = x[xrow];

    #pragma unroll 1
    for (int t = 0; t < RNN_T; ++t) {
        const _Float16* cbuf = hbuf[t & 1];
        _Float16*       nbuf = hbuf[(t & 1) ^ 1];

        // Consume the value prefetched last iteration; immediately issue the
        // next prefetch (clamped so the last row never reads past x).
        const float xcur = xpref;
        const int   tnxt = (t + 1 < RNN_T) ? (t + 1) : (RNN_T - 1);
        xpref = x[xrow + tnxt];

        // Redistribute x to (m = j + 8*half) slots up front: the 8 bpermutes
        // overlap the A-operand ds_loads and the WMMAs instead of the VALU tail.
        float xv[8];
        #pragma unroll
        for (int j = 0; j < 8; ++j)
            xv[j] = __shfl(xcur, j + 8 * half, 32);   // lane m loaded x[b0+m, t]

        // ---- A operands: h tile from LDS, ISA 16-bit A layout ----
        // lane row M = lq; K-runs of 8 halves at {K0 + half*8, +16}.
        const int abase0 = lq * ROWP + half * 8;          // K chunk 0 (K=0..31)
        v8h a0lo = *(const v8h*)(cbuf + abase0);
        v8h a0hi = *(const v8h*)(cbuf + abase0 + 16);
        const int abase1 = abase0 + 32;                   // K chunk 1 (K=32..63)
        v8h a1lo = *(const v8h*)(cbuf + abase1);
        v8h a1hi = *(const v8h*)(cbuf + abase1 + 16);
        v16h a0, a1;
        #pragma unroll
        for (int i = 0; i < 8; ++i) { a0[i] = a0lo[i]; a0[8 + i] = a0hi[i]; }
        #pragma unroll
        for (int i = 0; i < 8; ++i) { a1[i] = a1lo[i]; a1[8 + i] = a1hi[i]; }

        // ---- gh = h @ w_hh^T for this wave's 16 channels of r/z/n ----
        v8f accr = __builtin_amdgcn_wmma_f32_16x16x32_f16(false, a0, false, bm[0][0], (short)0, zero, false, false);
        v8f accz = __builtin_amdgcn_wmma_f32_16x16x32_f16(false, a0, false, bm[1][0], (short)0, zero, false, false);
        v8f accn = __builtin_amdgcn_wmma_f32_16x16x32_f16(false, a0, false, bm[2][0], (short)0, zero, false, false);
        accr = __builtin_amdgcn_wmma_f32_16x16x32_f16(false, a1, false, bm[0][1], (short)0, accr, false, false);
        accz = __builtin_amdgcn_wmma_f32_16x16x32_f16(false, a1, false, bm[1][1], (short)0, accz, false, false);
        accn = __builtin_amdgcn_wmma_f32_16x16x32_f16(false, a1, false, bm[2][1], (short)0, accn, false, false);

        // ---- gates + Euler update; lane element j is (m = j + 8*half, ch) ----
        #pragma unroll
        for (int j = 0; j < 8; ++j) {
            const int m = j + half * 8;

            const float r  = fast_sigmoid(xv[j] * wir + br + accr[j]);
            const float z  = fast_sigmoid(xv[j] * wiz + bz + accz[j]);
            const float hn = accn[j] + bhhn;
            const float n  = fast_tanh(xv[j] * win + bihn + r * hn);

            const float h    = hreg[j];
            const float hnew = h + EULER_K * (1.0f - z) * (n - h);
            hreg[j] = hnew;

            nbuf[m * ROWP + ch] = (_Float16)hnew;
            states[((size_t)(b0 + m) * RNN_T + t) * RNN_H + ch] = hnew;
        }
        __syncthreads();
    }
}

// out[b,t] = states[b,t,:] . w_out + b_out + x[b,t].  Wave-cooperative:
// half-wave per row, float4 per lane, shfl_xor tree reduce within 16 lanes.
// states (128 MB) is L2-resident (192 MB) right after the scan kernel.
__global__ void __launch_bounds__(256)
out_proj_kernel(const float* __restrict__ states,
                const float* __restrict__ x,
                const float* __restrict__ w_out,
                const float* __restrict__ b_out,
                float* __restrict__ out,
                int nrows)
{
    const int gwave = (blockIdx.x * blockDim.x + threadIdx.x) >> 5;
    const int lane  = threadIdx.x & 31;
    const int half  = lane >> 4;
    const int lq    = lane & 15;
    const int row   = gwave * 2 + half;
    if (row >= nrows) return;

    const float4 sv = *(const float4*)(states + (size_t)row * RNN_H + lq * 4);
    const float4 wv = *(const float4*)(w_out + lq * 4);
    float p = sv.x * wv.x + sv.y * wv.y + sv.z * wv.z + sv.w * wv.w;
    p += __shfl_xor(p, 1, 32);
    p += __shfl_xor(p, 2, 32);
    p += __shfl_xor(p, 4, 32);
    p += __shfl_xor(p, 8, 32);   // masks < 16 stay within each half-wave
    if (lq == 0) out[row] = p + b_out[0] + x[row];
}

extern "C" void kernel_launch(void* const* d_in, const int* in_sizes, int n_in,
                              void* d_out, int out_size, void* d_ws, size_t ws_size,
                              hipStream_t stream) {
    (void)in_sizes; (void)n_in; (void)out_size; (void)d_ws; (void)ws_size;
    const float* x     = (const float*)d_in[0];
    const float* w_ih  = (const float*)d_in[1];
    const float* w_hh  = (const float*)d_in[2];
    const float* b_ih  = (const float*)d_in[3];
    const float* b_hh  = (const float*)d_in[4];
    const float* w_out = (const float*)d_in[5];
    const float* b_out = (const float*)d_in[6];

    float* out    = (float*)d_out;                       // [B,T,1] first
    float* states = out + (size_t)RNN_B * RNN_T;         // then [B,T,H]

    // Serial scan: 2 blocks (batch tiles of 16), 4 waves each.
    gru_scan_kernel<<<dim3(RNN_B / 16), dim3(128), 0, stream>>>(
        x, w_ih, w_hh, b_ih, b_hh, states);

    // Output projection: nrows/2 waves, 8 waves per block.
    const int nrows  = RNN_B * RNN_T;                    // 524288
    const int nblk   = (nrows / 2 * 32) / 256;           // exact: 32768
    out_proj_kernel<<<dim3(nblk), dim3(256), 0, stream>>>(
        states, x, w_out, b_out, out, nrows);
}
